// self_SP_Block_807453852293
// MI455X (gfx1250) — compile-verified, matrix operand order checked by reading, not run on hardware
//
#include <hip/hip_runtime.h>

// ---------------- types & helpers ----------------
typedef __attribute__((ext_vector_type(16))) __bf16    v16bf;
typedef __attribute__((ext_vector_type(8)))  float     v8f;
typedef __attribute__((ext_vector_type(4)))  unsigned  u4v;
typedef __attribute__((ext_vector_type(4)))  float     f4v;

union Frag16 { v16bf v; u4v q[2]; };

__device__ __forceinline__ unsigned short f2bf(float f) {
  unsigned int u = __float_as_uint(f);
  u = u + 0x7FFFu + ((u >> 16) & 1u);          // round-to-nearest-even
  return (unsigned short)(u >> 16);
}
__device__ __forceinline__ unsigned pack2(unsigned short lo, unsigned short hi) {
  return (unsigned)lo | ((unsigned)hi << 16);
}
// monotone float->uint map so unsigned atomicMax == float max
__device__ __forceinline__ unsigned fmap(float f) {
  unsigned u = __float_as_uint(f);
  return (u & 0x80000000u) ? ~u : (u | 0x80000000u);
}
__device__ __forceinline__ float funmap(unsigned u) {
  unsigned v = (u & 0x80000000u) ? (u & 0x7FFFFFFFu) : ~u;
  return __uint_as_float(v);
}

// CDNA5 async global->LDS copy (ASYNCcnt path, no VGPR staging)
__device__ __forceinline__ void async_load_b128(unsigned lds_off, const void* gaddr) {
  asm volatile("global_load_async_to_lds_b128 %0, %1, off"
               :: "v"(lds_off), "v"((unsigned long long)(uintptr_t)gaddr)
               : "memory");
}
__device__ __forceinline__ void wait_async0() {
  asm volatile("s_wait_asynccnt 0x0" ::: "memory");
}

#define NEG_BIG (-3.402823466e38f)

// ---------------- prepass: NCHW f32 -> NHWC bf16 via LDS transpose ----------------
// block = (b, h, 64-wide W segment); coalesced b128 reads along W, b128 writes along C.
__global__ __launch_bounds__(256) void k_convert_x(const float* __restrict__ x,
                                                   unsigned* __restrict__ xbf) {
  __shared__ unsigned tl[64 * 65];        // [w 64][cpair 64], pad stride 65 (bank-safe)
  int bid  = blockIdx.x;
  int wseg = bid & 3;
  int h    = (bid >> 2) & 255;
  int b    = bid >> 10;
  int w0   = wseg * 64;
  int t    = threadIdx.x;

  // phase 1: read float4 along W for channel pair (2c2, 2c2+1), pack -> tl[w][c2]
  const f4v* xq = (const f4v*)x;
#pragma unroll
  for (int it = 0; it < 4; ++it) {
    int idx = it * 256 + t;               // 1024 tasks
    int wq  = idx & 15;                   // float4 index within 64 W
    int c2  = idx >> 4;                   // channel pair 0..63
    size_t base = (size_t)(b * 128 + 2 * c2) * 16384 + h * 64 + wseg * 16 + wq;
    f4v a = xq[base];
    f4v c = xq[base + 16384];
    int wb = wq * 4;
    tl[(wb + 0) * 65 + c2] = pack2(f2bf(a.x), f2bf(c.x));
    tl[(wb + 1) * 65 + c2] = pack2(f2bf(a.y), f2bf(c.y));
    tl[(wb + 2) * 65 + c2] = pack2(f2bf(a.z), f2bf(c.z));
    tl[(wb + 3) * 65 + c2] = pack2(f2bf(a.w), f2bf(c.w));
  }
  __syncthreads();

  // phase 2: contiguous b128 writes (8 channels per lane)
#pragma unroll
  for (int it = 0; it < 4; ++it) {
    int idx = it * 256 + t;               // 1024 uint4 outputs
    int cq  = idx & 15;
    int w   = idx >> 4;
    u4v v;
    v.x = tl[w * 65 + cq * 4 + 0];
    v.y = tl[w * 65 + cq * 4 + 1];
    v.z = tl[w * 65 + cq * 4 + 2];
    v.w = tl[w * 65 + cq * 4 + 3];
    ((u4v*)xbf)[(size_t)((b * 256 + h) * 256 + w0 + w) * 16 + cq] = v;
  }
}

// OIHW f32 -> [tap][O][I] bf16. 73,728 dwords per weight tensor.
__global__ __launch_bounds__(256) void k_prep_w(const float* __restrict__ src,
                                                unsigned* __restrict__ dst) {
  int d   = blockIdx.x * blockDim.x + threadIdx.x;
  int i2  = (d & 63) * 2;
  int o   = (d >> 6) & 127;
  int tap = d >> 13;
  float v0 = src[(size_t)(o * 128 + i2)     * 9 + tap];
  float v1 = src[(size_t)(o * 128 + i2 + 1) * 9 + tap];
  dst[d] = pack2(f2bf(v0), f2bf(v1));
}

// down_w [8][16][128] f32 (flat == [ko][c]) -> bf16 dwords. 8192 dwords.
__global__ __launch_bounds__(256) void k_prep_wd(const float* __restrict__ src,
                                                 unsigned* __restrict__ dst) {
  int d = blockIdx.x * blockDim.x + threadIdx.x;
  dst[d] = pack2(f2bf(src[2 * d]), f2bf(src[2 * d + 1]));
}

// fold bias+BN into per-channel scale/shift for both convs
__global__ void k_prep_bn(const float* c1b, const float* g1, const float* b1,
                          const float* m1, const float* v1,
                          const float* c2b, const float* g2, const float* b2,
                          const float* m2, const float* v2,
                          float* sc1, float* sh1, float* sc2, float* sh2) {
  int t = threadIdx.x;
  int c = t & 127;
  if (t < 128) {
    float s = g1[c] * rsqrtf(v1[c] + 1e-5f);
    sc1[c] = s; sh1[c] = (c1b[c] - m1[c]) * s + b1[c];
  } else {
    float s = g2[c] * rsqrtf(v2[c] + 1e-5f);
    sc2[c] = s; sh2[c] = (c2b[c] - m2[c]) * s + b2[c];
  }
}

__global__ __launch_bounds__(256) void k_init_colmax(unsigned* __restrict__ colu) {
  int i = blockIdx.x * blockDim.x + threadIdx.x;   // 131072
  colu[i] = fmap(NEG_BIG);
}

// ---------------- fused conv3x3 + BN + ReLU (implicit GEMM, WMMA bf16) ----------------
// Block: (b, h, 64-wide W segment); 8 waves, wave w owns out-channels 16w..16w+15,
// each wave runs 4 N-subtiles per A-fragment load (amortizes weight L2 traffic).
// Halo tile staged with global_load_async_to_lds_b128.
__global__ __launch_bounds__(256) void k_conv3x3(const unsigned* __restrict__ inq,
                                                 const unsigned* __restrict__ wq,
                                                 const float* __restrict__ sc,
                                                 const float* __restrict__ sh,
                                                 unsigned* __restrict__ outq) {
  __shared__ u4v lds4[3 * 66 * 16];   // [row 3][w 66][ch-quad 16] bf16, 50688 B

  int bid  = blockIdx.x;
  int wseg = bid & 3;
  int h    = (bid >> 2) & 255;
  int b    = bid >> 10;
  int w0   = wseg * 64;
  int t    = threadIdx.x;

  bool edge = (h == 0) | (h == 255) | (wseg == 0) | (wseg == 3);
  if (edge) {
    u4v z = {0u, 0u, 0u, 0u};
    for (int d = t; d < 3 * 66 * 16; d += 256) lds4[d] = z;
    __syncthreads();
  }
  // async halo fill: rows h-1..h+1, cols w0-1..w0+64, all 128 channels
  for (int d = t; d < 3 * 66 * 16; d += 256) {
    int cq = d & 15;
    int wi = (d >> 4) % 66;
    int r  = d / (66 * 16);
    int hh = h + r - 1;
    int ww = w0 + wi - 1;
    if (hh >= 0 && hh < 256 && ww >= 0 && ww < 256) {
      const u4v* g = (const u4v*)inq + (size_t)((b * 256 + hh) * 256 + ww) * 16 + cq;
      async_load_b128((unsigned)(uintptr_t)&lds4[d], g);
    }
  }
  wait_async0();
  __syncthreads();

  int lane = t & 31;
  int wave = t >> 5;
  int half = lane >> 4;
  int ln   = lane & 15;
  int o0   = wave * 16;

  v8f acc[4];
#pragma unroll
  for (int ns = 0; ns < 4; ++ns)
    acc[ns] = (v8f){0.f, 0.f, 0.f, 0.f, 0.f, 0.f, 0.f, 0.f};

  for (int tap = 0; tap < 9; ++tap) {
    int ky = tap / 3, kx = tap % 3;
    // A rows live at [tap][o0+ln][*]: 64 dwords = 16 quads per row
    const u4v* aw = (const u4v*)wq + (size_t)(tap * 128 + o0 + ln) * 16;
#pragma unroll
    for (int cc = 0; cc < 4; ++cc) {
      Frag16 a;
      a.q[0] = aw[cc * 4 + half];
      a.q[1] = aw[cc * 4 + 2 + half];
#pragma unroll
      for (int ns = 0; ns < 4; ++ns) {
        const u4v* l = lds4 + (ky * 66 + ns * 16 + ln + kx) * 16;
        Frag16 bb;
        bb.q[0] = l[cc * 4 + half];
        bb.q[1] = l[cc * 4 + 2 + half];
        acc[ns] = __builtin_amdgcn_wmma_f32_16x16x32_bf16(false, a.v, false, bb.v,
                                                          (short)0, acc[ns], false, false);
      }
    }
  }

  // epilogue: folded bias+BN, ReLU, bf16, one b128 store per subtile
  float scl[8], shf[8];
#pragma unroll
  for (int r = 0; r < 8; ++r) {
    int ch = o0 + r + half * 8;
    scl[r] = sc[ch]; shf[r] = sh[ch];
  }
#pragma unroll
  for (int ns = 0; ns < 4; ++ns) {
    float v[8];
#pragma unroll
    for (int r = 0; r < 8; ++r)
      v[r] = fmaxf(acc[ns][r] * scl[r] + shf[r], 0.f);
    u4v o;
    o.x = pack2(f2bf(v[0]), f2bf(v[1]));
    o.y = pack2(f2bf(v[2]), f2bf(v[3]));
    o.z = pack2(f2bf(v[4]), f2bf(v[5]));
    o.w = pack2(f2bf(v[6]), f2bf(v[7]));
    int w = w0 + ns * 16 + ln;
    ((u4v*)outq)[(size_t)((b * 256 + h) * 256 + w) * 16 + wave * 2 + half] = o;
  }
}

// ---------------- 1x1 down conv + row/col max pooling ----------------
__global__ __launch_bounds__(256) void k_downpool(const unsigned* __restrict__ hq,
                                                  const unsigned* __restrict__ wdq,
                                                  float* __restrict__ rowmax,
                                                  unsigned* __restrict__ colmaxu) {
  int bid = blockIdx.x;
  int h = bid & 255;
  int b = bid >> 8;
  int lane = threadIdx.x & 31;
  int wave = threadIdx.x >> 5;
  int half = lane >> 4;
  int ln   = lane & 15;
  int o0   = wave * 16;

  // hoist A fragments (weights invariant across the row)
  const u4v* aw = (const u4v*)wdq + (size_t)(o0 + ln) * 16;
  Frag16 afr[4];
#pragma unroll
  for (int cc = 0; cc < 4; ++cc) {
    afr[cc].q[0] = aw[cc * 4 + half];
    afr[cc].q[1] = aw[cc * 4 + 2 + half];
  }

  v8f tmax = {NEG_BIG, NEG_BIG, NEG_BIG, NEG_BIG, NEG_BIG, NEG_BIG, NEG_BIG, NEG_BIG};
  size_t rowpix = (size_t)((b * 256 + h) * 256);

  for (int nt = 0; nt < 16; ++nt) {
    int w = nt * 16 + ln;
    const u4v* bp = (const u4v*)hq + (rowpix + w) * 16;
    v8f a8 = {0.f, 0.f, 0.f, 0.f, 0.f, 0.f, 0.f, 0.f};
#pragma unroll
    for (int cc = 0; cc < 4; ++cc) {
      Frag16 bb;
      bb.q[0] = bp[cc * 4 + half];
      bb.q[1] = bp[cc * 4 + 2 + half];
      a8 = __builtin_amdgcn_wmma_f32_16x16x32_bf16(false, afr[cc].v, false, bb.v,
                                                   (short)0, a8, false, false);
    }
#pragma unroll
    for (int r = 0; r < 8; ++r) {
      int ko = o0 + r + half * 8;
      atomicMax(&colmaxu[(size_t)(b * 128 + ko) * 256 + w], fmap(a8[r]));
      tmax[r] = fmaxf(tmax[r], a8[r]);
    }
  }

  // rowmax: reduce over the 16 lanes of each half
#pragma unroll
  for (int r = 0; r < 8; ++r) {
    float v = tmax[r];
    v = fmaxf(v, __shfl_xor(v, 1, 32));
    v = fmaxf(v, __shfl_xor(v, 2, 32));
    v = fmaxf(v, __shfl_xor(v, 4, 32));
    v = fmaxf(v, __shfl_xor(v, 8, 32));
    if (ln == 0) {
      int ko = o0 + r + half * 8;
      rowmax[(size_t)(b * 128 + ko) * 256 + h] = v;
    }
  }
}

// ---------------- tiny reductions for out_add1 ----------------
__global__ void k_reduce8(const float* __restrict__ rowmax,
                          const unsigned* __restrict__ colmaxu,
                          const float* __restrict__ d8,
                          float* __restrict__ R8, float* __restrict__ C8) {
  int t = blockIdx.x * blockDim.x + threadIdx.x;   // 2048
  int bh = t & 1023;
  int b = bh >> 8, p = bh & 255;
  float s = 0.f;
  if (t < 1024) {
    for (int ko = 0; ko < 128; ++ko)
      s += rowmax[(size_t)(b * 128 + ko) * 256 + p] * d8[ko & 15];
    R8[bh] = s;
  } else {
    for (int ko = 0; ko < 128; ++ko)
      s += funmap(colmaxu[(size_t)(b * 128 + ko) * 256 + p]) * d8[ko & 15];
    C8[bh] = s;
  }
}

// ---------------- broadcast writers (bandwidth bound, b128 stores) ----------------
__global__ __launch_bounds__(256) void k_outcat(const float* __restrict__ rowmax,
                                                const unsigned* __restrict__ colmaxu,
                                                float* __restrict__ out) {
  int bid = blockIdx.x;                       // (b*128+ko)*64 + h/4  : 32768 blocks
  int h   = (bid & 63) * 4 + (threadIdx.x >> 6);
  int bko = bid >> 6;
  int t   = threadIdx.x & 63;
  float rv = rowmax[(size_t)bko * 256 + h];
  u4v cu = ((const u4v*)(colmaxu + (size_t)bko * 256))[t];
  f4v o;
  o.x = rv + funmap(cu.x); o.y = rv + funmap(cu.y);
  o.z = rv + funmap(cu.z); o.w = rv + funmap(cu.w);
  ((f4v*)out)[((size_t)bko * 256 + h) * 64 + t] = o;
}

__global__ __launch_bounds__(256) void k_outadd(const float* __restrict__ R8,
                                                const float* __restrict__ C8,
                                                float* __restrict__ out) {
  int bh = blockIdx.x * 4 + (threadIdx.x >> 6);   // 256 blocks -> 1024 (b,h)
  int t  = threadIdx.x & 63;
  int b  = bh >> 8;
  float rv = R8[bh];
  f4v cv = ((const f4v*)(C8 + b * 256))[t];
  f4v o = {rv + cv.x, rv + cv.y, rv + cv.z, rv + cv.w};
  ((f4v*)out)[(size_t)bh * 64 + t] = o;
}

// ---------------- launch ----------------
extern "C" void kernel_launch(void* const* d_in, const int* in_sizes, int n_in,
                              void* d_out, int out_size, void* d_ws, size_t ws_size,
                              hipStream_t stream) {
  const float* x   = (const float*)d_in[0];
  const float* c1w = (const float*)d_in[1];
  const float* c1b = (const float*)d_in[2];
  const float* g1  = (const float*)d_in[3];
  const float* b1  = (const float*)d_in[4];
  const float* m1  = (const float*)d_in[5];
  const float* v1  = (const float*)d_in[6];
  const float* c2w = (const float*)d_in[7];
  const float* c2b = (const float*)d_in[8];
  const float* g2  = (const float*)d_in[9];
  const float* b2  = (const float*)d_in[10];
  const float* m2  = (const float*)d_in[11];
  const float* v2  = (const float*)d_in[12];
  const float* dw  = (const float*)d_in[13];
  const float* d8  = (const float*)d_in[14];
  float* out = (float*)d_out;

  char* ws = (char*)d_ws;
  size_t off = 0;
  auto take = [&](size_t bytes) {
    void* p = ws + off;
    off = (off + bytes + 255) & ~(size_t)255;
    return p;
  };
  unsigned* xbf = (unsigned*)take(67108864);  // x NHWC bf16; reused as h2
  unsigned* h1  = (unsigned*)take(67108864);
  unsigned* w1t = (unsigned*)take(294912);
  unsigned* w2t = (unsigned*)take(294912);
  unsigned* wdt = (unsigned*)take(32768);
  float* sc1 = (float*)take(512);
  float* sh1 = (float*)take(512);
  float* sc2 = (float*)take(512);
  float* sh2 = (float*)take(512);
  float*    rowmax = (float*)take(524288);
  unsigned* colu   = (unsigned*)take(524288);
  float* R8 = (float*)take(4096);
  float* C8 = (float*)take(4096);

  k_convert_x<<<4096, 256, 0, stream>>>(x, xbf);
  k_prep_w<<<288, 256, 0, stream>>>(c1w, w1t);
  k_prep_w<<<288, 256, 0, stream>>>(c2w, w2t);
  k_prep_wd<<<32, 256, 0, stream>>>(dw, wdt);
  k_prep_bn<<<1, 256, 0, stream>>>(c1b, g1, b1, m1, v1,
                                   c2b, g2, b2, m2, v2,
                                   sc1, sh1, sc2, sh2);
  k_init_colmax<<<512, 256, 0, stream>>>(colu);

  k_conv3x3<<<4096, 256, 0, stream>>>(xbf, w1t, sc1, sh1, h1);
  k_conv3x3<<<4096, 256, 0, stream>>>(h1, w2t, sc2, sh2, xbf);   // h2 aliases xbf

  k_downpool<<<1024, 256, 0, stream>>>(xbf, wdt, rowmax, colu);
  k_reduce8<<<8, 256, 0, stream>>>(rowmax, colu, d8, R8, C8);

  k_outcat<<<32768, 256, 0, stream>>>(rowmax, colu, out);
  k_outadd<<<256, 256, 0, stream>>>(R8, C8, out + 33554432);
}
